// EncoderBlock_12275016532287
// MI455X (gfx1250) — compile-verified, matrix operand order checked by reading, not run on hardware
//
#include <hip/hip_runtime.h>
#include <hip/hip_bf16.h>
#include <math.h>

// ---------------------------------------------------------------------------
// Types / helpers
// ---------------------------------------------------------------------------
typedef _Float16 v16h __attribute__((ext_vector_type(16)));
typedef _Float16 v8h  __attribute__((ext_vector_type(8)));
typedef float    v8f  __attribute__((ext_vector_type(8)));
typedef int      v4i  __attribute__((vector_size(16)));   // matches builtin param

#define WMMA_F32_F16(a, b, c) \
  __builtin_amdgcn_wmma_f32_16x16x32_f16(false, (a), false, (b), (short)0, (c), false, false)

// ---------------------------------------------------------------------------
// CDNA5 async global->LDS copy (ASYNCcnt-tracked, no VGPR round trip).
// Signature probe-verified: (v4i AS(1)*, v4i AS(3)*, imm offset, imm cpol).
// ---------------------------------------------------------------------------
#if defined(__gfx1250__) && __has_builtin(__builtin_amdgcn_global_load_async_to_lds_b128)
#define HAVE_ASYNC_LDS 1
#else
#define HAVE_ASYNC_LDS 0
#endif

static __device__ __forceinline__ void async_copy_b128(void* lds, const void* g) {
#if HAVE_ASYNC_LDS
  // AS1 and flat share address values on AMDGPU -> inttoptr is safe for global.
  __attribute__((address_space(1))) v4i* gp =
      (__attribute__((address_space(1))) v4i*)(uintptr_t)g;
  // generic -> AS3 needs a real addrspacecast; do it, then bitcast pointee.
  __attribute__((address_space(3))) char* lc =
      (__attribute__((address_space(3))) char*)lds;
  __attribute__((address_space(3))) v4i* lp =
      (__attribute__((address_space(3))) v4i*)lc;
  __builtin_amdgcn_global_load_async_to_lds_b128(gp, lp, 0, 0);
#else
  *(v8h*)lds = *(const v8h*)g;
#endif
}

static __device__ __forceinline__ void wait_async0() {
#if HAVE_ASYNC_LDS
#if __has_builtin(__builtin_amdgcn_s_wait_asynccnt)
  __builtin_amdgcn_s_wait_asynccnt(0);
#else
  asm volatile("s_wait_asynccnt 0" ::: "memory");
#endif
#endif
}

// Build a 16-half WMMA A/B fragment for one lane from a row-major run.
// CDNA5 16-bit operand layout: lane-half base8 = (lane>=16)?8:0;
// halves 0..7 -> k = base8+0..7, halves 8..15 -> k = base8+16..23.
// Caller passes p already offset by base8; we read [p, p+8) and [p+16, p+24).
static __device__ __forceinline__ v16h load_frag(const _Float16* p) {
  v8h lo = *(const v8h*)(p);
  v8h hi = *(const v8h*)(p + 16);
  v16h f;
#pragma unroll
  for (int e = 0; e < 8; ++e) { f[e] = lo[e]; f[e + 8] = hi[e]; }
  return f;
}

// ---------------------------------------------------------------------------
// f32 -> f16 weight conversion
// ---------------------------------------------------------------------------
__global__ __launch_bounds__(256) void cvt_f16_kernel(
    const float* __restrict__ s, _Float16* __restrict__ d, int n) {
  int i = (blockIdx.x * 256 + threadIdx.x) * 4;
  if (i < n) {
    float4 v = *(const float4*)(s + i);
    d[i + 0] = (_Float16)v.x;
    d[i + 1] = (_Float16)v.y;
    d[i + 2] = (_Float16)v.z;
    d[i + 3] = (_Float16)v.w;
  }
}

// ---------------------------------------------------------------------------
// LayerNorm over C=1024, output f16; one block (256 thr, 8 waves) per row
// ---------------------------------------------------------------------------
__global__ __launch_bounds__(256) void ln_f16_kernel(
    const float* __restrict__ x, const float* __restrict__ g,
    const float* __restrict__ bt, _Float16* __restrict__ out) {
  const int C = 1024;
  size_t row = blockIdx.x;
  const float4 v = ((const float4*)(x + row * C))[threadIdx.x];
  float s  = v.x + v.y + v.z + v.w;
  float ss = v.x * v.x + v.y * v.y + v.z * v.z + v.w * v.w;
#pragma unroll
  for (int m = 16; m >= 1; m >>= 1) {
    s  += __shfl_xor(s,  m, 32);
    ss += __shfl_xor(ss, m, 32);
  }
  __shared__ float red[2][8];
  int wave = threadIdx.x >> 5, lane = threadIdx.x & 31;
  if (lane == 0) { red[0][wave] = s; red[1][wave] = ss; }
  __syncthreads();
  float S = 0.f, SS = 0.f;
#pragma unroll
  for (int i = 0; i < 8; ++i) { S += red[0][i]; SS += red[1][i]; }
  const float mu = S * (1.0f / 1024.0f);
  const float rs = rsqrtf(SS * (1.0f / 1024.0f) - mu * mu + 1e-5f);
  int c = threadIdx.x * 4;
  float4 gg = *(const float4*)(g + c);
  float4 bb = *(const float4*)(bt + c);
  _Float16* op = out + row * C + c;
  op[0] = (_Float16)((v.x - mu) * rs * gg.x + bb.x);
  op[1] = (_Float16)((v.y - mu) * rs * gg.y + bb.y);
  op[2] = (_Float16)((v.z - mu) * rs * gg.z + bb.z);
  op[3] = (_Float16)((v.w - mu) * rs * gg.w + bb.w);
}

// ---------------------------------------------------------------------------
// WMMA GEMM: C[M,N] = A[M,K](f16,rm) * B[K,N](f16,rm) (+bias)(+GELU)(+residual)
// 128x128x32 block tiles, 8 waves, wave tile 64x32 (4x2 WMMA tiles).
// Double-buffered LDS; A tile via async copy, B tile transposed via regs.
// Epilogue variants specialized at compile time (no per-element branches).
// ---------------------------------------------------------------------------
#define BM 128
#define BN 128
#define BK 32
#define LDT 40  // padded LDS tile row stride in halves (80B -> conflict-free)

template <bool HAS_BIAS, bool HAS_RES, bool GELU, bool OUT_F16>
__global__ __launch_bounds__(256) void gemm_wmma_kernel(
    const _Float16* __restrict__ A, const _Float16* __restrict__ B,
    const float* __restrict__ bias, const float* __restrict__ residual,
    float* __restrict__ outF, _Float16* __restrict__ outH,
    int M, int N, int K) {
  __shared__ _Float16 As[2][BM * LDT];   // [m][k] row-major
  __shared__ _Float16 Bs[2][BN * LDT];   // [n][k] transposed at store
  const int tid  = threadIdx.x;
  const int wave = tid >> 5, lane = tid & 31;
  const int r  = lane & 15;
  const int hb = (lane >> 4) << 3;       // k base-8 for operand halves
  const int mBlk = blockIdx.y * BM, nBlk = blockIdx.x * BN;
  const int mw = (wave & 1) * 64, nw = (wave >> 1) * 32;

  // staging coords: A = 512 b128 chunks (2/thread), B = 512 8-wide runs
  const int rowA0 = tid >> 2,           kcA0 = (tid & 3) << 3;
  const int rowA1 = (tid + 256) >> 2,   kcA1 = kcA0;
  const int krB0  = tid >> 4,           ncB  = (tid & 15) << 3;
  const int krB1  = krB0 + 16;
  const _Float16* gA0 = A + (size_t)(mBlk + rowA0) * K + kcA0;
  const _Float16* gA1 = A + (size_t)(mBlk + rowA1) * K + kcA1;
  const _Float16* gB0 = B + (size_t)krB0 * N + nBlk + ncB;
  const _Float16* gB1 = B + (size_t)krB1 * N + nBlk + ncB;

  v8h bv0, bv1;

  v8f acc[4][2] = {};

  const int nk = K / BK;
  // prologue: stage tile 0
  async_copy_b128(&As[0][rowA0 * LDT + kcA0], gA0);
  async_copy_b128(&As[0][rowA1 * LDT + kcA1], gA1);
  bv0 = *(const v8h*)(gB0);
  bv1 = *(const v8h*)(gB1);
#pragma unroll
  for (int j = 0; j < 8; ++j) Bs[0][(ncB + j) * LDT + krB0] = bv0[j];
#pragma unroll
  for (int j = 0; j < 8; ++j) Bs[0][(ncB + j) * LDT + krB1] = bv1[j];

  for (int kt = 0; kt < nk; ++kt) {
    const int cur = kt & 1;
    wait_async0();
    __syncthreads();   // tile 'cur' fully staged, everyone done with 'cur^1'

    // issue next tile: async A copies + B loads into regs (overlap compute)
    if (kt + 1 < nk) {
      const size_t ko = (size_t)(kt + 1) * BK;
      async_copy_b128(&As[cur ^ 1][rowA0 * LDT + kcA0], gA0 + ko);
      async_copy_b128(&As[cur ^ 1][rowA1 * LDT + kcA1], gA1 + ko);
      bv0 = *(const v8h*)(gB0 + ko * N);
      bv1 = *(const v8h*)(gB1 + ko * N);
    }

    v16h af[4], bf[2];
#pragma unroll
    for (int i = 0; i < 4; ++i)
      af[i] = load_frag(&As[cur][(mw + i * 16 + r) * LDT + hb]);
#pragma unroll
    for (int j = 0; j < 2; ++j)
      bf[j] = load_frag(&Bs[cur][(nw + j * 16 + r) * LDT + hb]);
#pragma unroll
    for (int i = 0; i < 4; ++i)
#pragma unroll
      for (int j = 0; j < 2; ++j)
        acc[i][j] = WMMA_F32_F16(af[i], bf[j], acc[i][j]);

    // transpose-scatter next B tile after the WMMAs (hides loadcnt wait)
    if (kt + 1 < nk) {
#pragma unroll
      for (int j = 0; j < 8; ++j) Bs[cur ^ 1][(ncB + j) * LDT + krB0] = bv0[j];
#pragma unroll
      for (int j = 0; j < 8; ++j) Bs[cur ^ 1][(ncB + j) * LDT + krB1] = bv1[j];
    }
  }

  // epilogue. C-layout: VGPR v -> row (v + 8*lanehi), lane&15 -> col.
  // Only 2 distinct output columns per lane -> hoist bias out of the loops.
  float bcol[2] = {0.f, 0.f};
  if (HAS_BIAS) {
#pragma unroll
    for (int j = 0; j < 2; ++j) bcol[j] = bias[(size_t)(nBlk + nw + j * 16 + r)];
  }
#pragma unroll
  for (int i = 0; i < 4; ++i) {
#pragma unroll
    for (int j = 0; j < 2; ++j) {
#pragma unroll
      for (int v = 0; v < 8; ++v) {
        int ml = mw + i * 16 + v + hb;   // hb == 8*lanehi
        int nl = nw + j * 16 + r;
        size_t row = (size_t)(mBlk + ml);
        size_t col = (size_t)(nBlk + nl);
        float c = acc[i][j][v];
        if (HAS_BIAS) c += bcol[j];
        if (GELU)     c  = 0.5f * c * (1.0f + erff(c * 0.70710678118654752f));
        if (HAS_RES)  c += residual[row * (size_t)N + col];
        if (OUT_F16)  outH[row * (size_t)N + col] = (_Float16)c;
        else          outF[row * (size_t)N + col] = c;
      }
    }
  }
}

// ---------------------------------------------------------------------------
// Flash attention (online softmax), WMMA f16, double-buffered K/V tiles.
// qkv layout: [b*2048+n][3072], q at col h*64+d, k at +1024, v at +2048.
// grid = B*H*(N/64) blocks, 128 threads = 4 waves; wave = 16 query rows.
// Row-sum of P computed on the matrix pipe (P @ ones) instead of shuffles.
// ---------------------------------------------------------------------------
#define KPAD 72   // K tile row stride (halves), conflict-padded
#define VPAD 40   // V^T / P tile row stride (halves)

__global__ __launch_bounds__(128) void flash_attn_kernel(
    const _Float16* __restrict__ qkv, _Float16* __restrict__ o16) {
  const int N = 2048, LDQ = 3072;
  int qt = blockIdx.x & 31;          // query tile (64 rows each)
  int bh = blockIdx.x >> 5;
  int h = bh & 15, b = bh >> 4;

  __shared__ _Float16 Ks[2][32 * KPAD];     // [key][d]
  __shared__ _Float16 Vt[2][64 * VPAD];     // [d][key]  (transposed)
  __shared__ _Float16 Ps[4 * 16 * VPAD];    // per-wave P 16x32

  const int tid = threadIdx.x, wave = tid >> 5, lane = tid & 31;
  const int r = lane & 15, hb = (lane >> 4) << 3;
  const _Float16* Qb = qkv + (size_t)b * N * LDQ + h * 64;
  const _Float16* Kb = Qb + 1024;
  const _Float16* Vb = Qb + 2048;
  const int q0 = qt * 64 + wave * 16;

  v16h aq[2];
#pragma unroll
  for (int dc = 0; dc < 2; ++dc)
    aq[dc] = load_frag(Qb + (size_t)(q0 + r) * LDQ + dc * 32 + hb);

  // constant all-ones B fragment: P @ ones -> per-row sums (layout-agnostic)
  v16h ones;
#pragma unroll
  for (int e = 0; e < 16; ++e) ones[e] = (_Float16)1.0f;

  v8f o[4] = {};
  float mrow[8], lrow[8];
#pragma unroll
  for (int v = 0; v < 8; ++v) { mrow[v] = -1e30f; lrow[v] = 0.f; }

  _Float16* Pw = &Ps[wave * 16 * VPAD];

  // staging coords: 32 keys x 64 halves = 256 b128 chunks (2/thread)
  const int kr0 = tid >> 3,        c0 = (tid & 7) << 3;
  const int kr1 = kr0 + 16,        c1 = c0;
  const _Float16* gK0 = Kb + (size_t)kr0 * LDQ + c0;
  const _Float16* gK1 = Kb + (size_t)kr1 * LDQ + c1;
  const _Float16* gV0 = Vb + (size_t)kr0 * LDQ + c0;
  const _Float16* gV1 = Vb + (size_t)kr1 * LDQ + c1;

  v8h vv0, vv1;

  const int NB = N / 32;
  // prologue: stage block 0
  async_copy_b128(&Ks[0][kr0 * KPAD + c0], gK0);
  async_copy_b128(&Ks[0][kr1 * KPAD + c1], gK1);
  vv0 = *(const v8h*)(gV0);
  vv1 = *(const v8h*)(gV1);
#pragma unroll
  for (int j = 0; j < 8; ++j) Vt[0][(c0 + j) * VPAD + kr0] = vv0[j];
#pragma unroll
  for (int j = 0; j < 8; ++j) Vt[0][(c1 + j) * VPAD + kr1] = vv1[j];

  for (int kb = 0; kb < NB; ++kb) {
    const int cur = kb & 1;
    wait_async0();
    __syncthreads();

    if (kb + 1 < NB) {
      const size_t o_ = (size_t)((kb + 1) * 32) * LDQ;
      async_copy_b128(&Ks[cur ^ 1][kr0 * KPAD + c0], gK0 + o_);
      async_copy_b128(&Ks[cur ^ 1][kr1 * KPAD + c1], gK1 + o_);
      vv0 = *(const v8h*)(gV0 + o_);
      vv1 = *(const v8h*)(gV1 + o_);
    }

    // S(16x32) = Q(16x64) . K^T : B-frag lane -> key, halves -> d
    v8f s[2] = {};
#pragma unroll
    for (int kt = 0; kt < 2; ++kt)
#pragma unroll
      for (int dc = 0; dc < 2; ++dc) {
        v16h bk = load_frag(&Ks[cur][(kt * 16 + r) * KPAD + dc * 32 + hb]);
        s[kt] = WMMA_F32_F16(aq[dc], bk, s[kt]);
      }

    // online softmax max-pass; row = v + 8*lanehi, reduce across 16-lane half
    float alpha[8];
#pragma unroll
    for (int v = 0; v < 8; ++v) {
      float s0 = s[0][v] * 0.125f, s1 = s[1][v] * 0.125f;  // * D^-0.5
      float mx = fmaxf(s0, s1);
#pragma unroll
      for (int msk = 8; msk >= 1; msk >>= 1) mx = fmaxf(mx, __shfl_xor(mx, msk, 32));
      float mnew = fmaxf(mrow[v], mx);
      alpha[v] = __expf(mrow[v] - mnew);
      mrow[v]  = mnew;
      s[0][v] = __expf(s0 - mnew);
      s[1][v] = __expf(s1 - mnew);
    }

    // C-layout -> A-fragment layout via per-wave LDS bounce (in-order DS)
#pragma unroll
    for (int kt = 0; kt < 2; ++kt)
#pragma unroll
      for (int v = 0; v < 8; ++v)
        Pw[(v + hb) * VPAD + kt * 16 + r] = (_Float16)s[kt][v];

    v16h pf = load_frag(&Pw[r * VPAD + hb]);

    // row sums on the matrix pipe: P(16x32) @ ones(32x16); every column of
    // the C tile holds sum_k P[row,k], same row<->(v,lanehalf) map as lrow.
    v8f zero = {};
    v8f rsum = WMMA_F32_F16(pf, ones, zero);
#pragma unroll
    for (int v = 0; v < 8; ++v) lrow[v] = lrow[v] * alpha[v] + rsum[v];

    // O(16x64) = alpha*O + P(16x32) . V(32x64)
#pragma unroll
    for (int t = 0; t < 4; ++t) {
#pragma unroll
      for (int v = 0; v < 8; ++v) o[t][v] *= alpha[v];
      v16h bv = load_frag(&Vt[cur][(t * 16 + r) * VPAD + hb]);
      o[t] = WMMA_F32_F16(pf, bv, o[t]);
    }

    // scatter next V tile after the WMMAs (hides loadcnt wait)
    if (kb + 1 < NB) {
#pragma unroll
      for (int j = 0; j < 8; ++j) Vt[cur ^ 1][(c0 + j) * VPAD + kr0] = vv0[j];
#pragma unroll
      for (int j = 0; j < 8; ++j) Vt[cur ^ 1][(c1 + j) * VPAD + kr1] = vv1[j];
    }
  }

  // write o16[b*2048+n][h*64+d] = O / l
#pragma unroll
  for (int t = 0; t < 4; ++t)
#pragma unroll
    for (int v = 0; v < 8; ++v) {
      float val = o[t][v] / lrow[v];
      o16[(size_t)(b * N + q0 + v + hb) * 1024 + h * 64 + t * 16 + r] = (_Float16)val;
    }
}

// ---------------------------------------------------------------------------
// Launch: LN1 -> QKV GEMM -> flash attn -> proj(+res) -> LN2 -> FC1(GELU) -> FC2(+res)
// ---------------------------------------------------------------------------
extern "C" void kernel_launch(void* const* d_in, const int* in_sizes, int n_in,
                              void* d_out, int out_size, void* d_ws, size_t ws_size,
                              hipStream_t stream) {
  (void)in_sizes; (void)n_in; (void)out_size; (void)ws_size;
  const float* x     = (const float*)d_in[0];
  const float* ln1g  = (const float*)d_in[1];
  const float* ln1b  = (const float*)d_in[2];
  const float* w_qkv = (const float*)d_in[3];
  const float* w_prj = (const float*)d_in[4];
  const float* b_prj = (const float*)d_in[5];
  const float* ln2g  = (const float*)d_in[6];
  const float* ln2b  = (const float*)d_in[7];
  const float* w_fc1 = (const float*)d_in[8];
  const float* b_fc1 = (const float*)d_in[9];
  const float* w_fc2 = (const float*)d_in[10];
  const float* b_fc2 = (const float*)d_in[11];
  float* out = (float*)d_out;

  const int DIM = 1024, NTOK = 4096, HID = 4096, NQ = 3072;

  char* ws = (char*)d_ws;
  size_t off = 0;
  auto alloc = [&](size_t bytes) {
    void* p = ws + off;
    off += (bytes + 255) & ~(size_t)255;
    return p;
  };
  _Float16* wqkv16 = (_Float16*)alloc((size_t)DIM * NQ * 2);
  _Float16* wprj16 = (_Float16*)alloc((size_t)DIM * DIM * 2);
  _Float16* wfc116 = (_Float16*)alloc((size_t)DIM * HID * 2);
  _Float16* wfc216 = (_Float16*)alloc((size_t)HID * DIM * 2);
  _Float16* h16    = (_Float16*)alloc((size_t)NTOK * DIM * 2);
  _Float16* qkv16  = (_Float16*)alloc((size_t)NTOK * NQ * 2);
  _Float16* o16    = (_Float16*)alloc((size_t)NTOK * DIM * 2);
  float*    x1     = (float*)   alloc((size_t)NTOK * DIM * 4);
  // g16 (4096x4096 f16 = 33.5MB) reuses the qkv16+o16 region (dead by then)
  _Float16* g16 = qkv16;

  // weights f32 -> f16 (once per launch; deterministic)
  cvt_f16_kernel<<<dim3(DIM * NQ  / 1024), 256, 0, stream>>>(w_qkv, wqkv16, DIM * NQ);
  cvt_f16_kernel<<<dim3(DIM * DIM / 1024), 256, 0, stream>>>(w_prj, wprj16, DIM * DIM);
  cvt_f16_kernel<<<dim3(DIM * HID / 1024), 256, 0, stream>>>(w_fc1, wfc116, DIM * HID);
  cvt_f16_kernel<<<dim3(HID * DIM / 1024), 256, 0, stream>>>(w_fc2, wfc216, HID * DIM);

  // h = LN1(x) (f16)
  ln_f16_kernel<<<dim3(NTOK), 256, 0, stream>>>(x, ln1g, ln1b, h16);

  // qkv = h @ Wqkv   (f16 out, attention-friendly layout)
  gemm_wmma_kernel<false, false, false, true>
      <<<dim3(NQ / 128, NTOK / 128), 256, 0, stream>>>(
          h16, wqkv16, nullptr, nullptr, nullptr, qkv16, NTOK, NQ, DIM);

  // o = softmax(q k^T / sqrt(64)) v     (flash, per 64-query tile)
  flash_attn_kernel<<<dim3(2 * 16 * (2048 / 64)), 128, 0, stream>>>(qkv16, o16);

  // x1 = x + o @ Wprj + b_prj           (f32 out)
  gemm_wmma_kernel<true, true, false, false>
      <<<dim3(DIM / 128, NTOK / 128), 256, 0, stream>>>(
          o16, wprj16, b_prj, x, x1, nullptr, NTOK, DIM, DIM);

  // h = LN2(x1) (f16)
  ln_f16_kernel<<<dim3(NTOK), 256, 0, stream>>>(x1, ln2g, ln2b, h16);

  // g = gelu(h @ Wfc1 + b_fc1)          (f16 out)
  gemm_wmma_kernel<true, false, true, true>
      <<<dim3(HID / 128, NTOK / 128), 256, 0, stream>>>(
          h16, wfc116, b_fc1, nullptr, nullptr, g16, NTOK, HID, DIM);

  // out = x1 + g @ Wfc2 + b_fc2         (f32 out)
  gemm_wmma_kernel<true, true, false, false>
      <<<dim3(DIM / 128, NTOK / 128), 256, 0, stream>>>(
          g16, wfc216, b_fc2, x1, out, nullptr, NTOK, DIM, HID);
}
